// Generator_20151986553429
// MI455X (gfx1250) — matrix operand order repaired and checked by reading, not compiled, not run on hardware
//
#include <hip/hip_runtime.h>
#include <hip/hip_bf16.h>
#include <math.h>

// ---------------- types ----------------
typedef __bf16 bf16_t;
typedef bf16_t v16bf __attribute__((ext_vector_type(16)));
typedef float  v8f   __attribute__((ext_vector_type(8)));
typedef unsigned int v4u __attribute__((ext_vector_type(4)));
typedef unsigned int v8u __attribute__((ext_vector_type(8)));

union FragB16 { v16bf v; bf16_t e[16]; };
union FragF32 { v8f   v; float  e[8];  };

#define S_     256
#define GC_    256
#define ZD_    128
#define NBLK_  8
#define BATCH_ 4
#define NPIX_  65536   // 256*256

__device__ __forceinline__ float leakyf(float v) { return v > 0.f ? v : 0.2f * v; }

__device__ __forceinline__ v8f wmma_bf16(v16bf a, v16bf b, v8f c) {
  // D = A(16x32 bf16) * B(32x16 bf16) + C(16x16 f32)
  return __builtin_amdgcn_wmma_f32_16x16x32_bf16(false, a, false, b, (short)0, c,
                                                 false, false);
}

// ---- fragment loaders (ISA 7.12.2 layouts, wave32) ----
// A (16x32, bf16): lane row M = lane%16; e<8 -> K=kk+8*half+e ; e>=8 -> K=kk+16+8*half+(e-8)
__device__ __forceinline__ v16bf loadA_bf16(const bf16_t* base, int row, int kk, int ld, int lane) {
  const bf16_t* p = base + row * ld;
  int h = (lane >> 4) << 3;
  FragB16 f;
#pragma unroll
  for (int e = 0; e < 8; ++e) f.e[e]     = p[kk + h + e];
#pragma unroll
  for (int e = 0; e < 8; ++e) f.e[8 + e] = p[kk + 16 + h + e];
  return f.v;
}
// B (32x16): lane col n = lane%16 (rowp already offset by col*ld); e<8 -> K=kk+8*half+e ; e>=8 -> K=kk+16+8*half+(e-8)
__device__ __forceinline__ v16bf loadB_bf16(const bf16_t* rowp, int kk, int lane) {
  int h = (lane >> 4) << 3;
  FragB16 f;
#pragma unroll
  for (int e = 0; e < 8; ++e) f.e[e]     = rowp[kk + h + e];
#pragma unroll
  for (int e = 0; e < 8; ++e) f.e[8 + e] = rowp[kk + 16 + h + e];
  return f.v;
}

// low 32 bits of a generic pointer into LDS == LDS byte offset (aperture rule)
__device__ __forceinline__ unsigned lds_off_of(const void* p) {
  return (unsigned)(unsigned long long)(uintptr_t)p;
}

// ================= prep: f32 -> bf16 bulk convert =================
__global__ __launch_bounds__(256) void k_cvt(const float* __restrict__ src,
                                             bf16_t* __restrict__ dst, int n) {
  int i = blockIdx.x * 256 + threadIdx.x;
  if (i < n) dst[i] = (bf16_t)src[i];
}

// ================= K1: superpixel mean pooling =================
__global__ __launch_bounds__(256) void k_pool(const float* __restrict__ images,
                                              const int* __restrict__ seg,
                                              float* __restrict__ feats) {
  __shared__ float lsum[S_ * 3];
  __shared__ float lcnt[S_];
  const int b = blockIdx.x, t = threadIdx.x;
  for (int i = t; i < S_ * 3; i += 256) lsum[i] = 0.f;
  lcnt[t] = 0.f;
  __syncthreads();
  const float* im = images + (size_t)b * 3 * NPIX_;
  const int*   sg = seg + (size_t)b * NPIX_;
  for (int it = 0; it < NPIX_ / 256; ++it) {
    int p = it * 256 + t;
    int s = sg[p];
    atomicAdd(&lsum[s * 3 + 0], im[0 * NPIX_ + p]);
    atomicAdd(&lsum[s * 3 + 1], im[1 * NPIX_ + p]);
    atomicAdd(&lsum[s * 3 + 2], im[2 * NPIX_ + p]);
    atomicAdd(&lcnt[s], 1.f);
  }
  __syncthreads();
  float inv = 1.f / (lcnt[t] + 1e-6f);
  feats[(b * S_ + t) * 3 + 0] = lsum[t * 3 + 0] * inv;
  feats[(b * S_ + t) * 3 + 1] = lsum[t * 3 + 1] * inv;
  feats[(b * S_ + t) * 3 + 2] = lsum[t * 3 + 2] * inv;
}

// ================= K2: fused proj + 8 GNN blocks, LDS-resident =================
// LDS: x[s][c] bf16 128KB | h[c][s] bf16 128KB | mean 1KB | s1 2KB | style 2KB
__global__ __launch_bounds__(256) void k_core(const float* __restrict__ z,
                                              const float* __restrict__ feats,
                                              const float* __restrict__ proj_w,
                                              const float* __restrict__ proj_b,
                                              const bf16_t* __restrict__ blkw_bf,
                                              const float* __restrict__ blkb,
                                              const float* __restrict__ a1w,
                                              const float* __restrict__ a1b,
                                              const float* __restrict__ a2w,
                                              const float* __restrict__ a2b,
                                              const bf16_t* __restrict__ Aadj_bf,
                                              bf16_t* __restrict__ xsup_out) {
  extern __shared__ char smem[];
  bf16_t* x_lds    = (bf16_t*)smem;                    // [S][GC] node-major
  bf16_t* h_lds    = (bf16_t*)(smem + 131072);         // [GC][S] channel-major
  float*  mean_lds = (float*)(smem + 262144);          // 256
  float*  s1_lds   = (float*)(smem + 263168);          // 512 (also caches z)
  float*  sty_lds  = (float*)(smem + 265216);          // 512 (gamma|beta)

  const int b = blockIdx.x;
  const int t = threadIdx.x;
  const int lane = t & 31, w = t >> 5;

  if (t < ZD_) s1_lds[t] = z[b * ZD_ + t];
  __syncthreads();

  // ---- proj: x[s=t][o] = sqrt(2/131)*(W @ [feat;z]) + b ----
  {
    float f0 = feats[(b * S_ + t) * 3 + 0];
    float f1 = feats[(b * S_ + t) * 3 + 1];
    float f2 = feats[(b * S_ + t) * 3 + 2];
    const float sc = sqrtf(2.f / 131.f);
    for (int o = 0; o < GC_; ++o) {
      const float* wr = proj_w + o * 131;
      float acc = wr[0] * f0 + wr[1] * f1 + wr[2] * f2;
      for (int c = 0; c < ZD_; ++c) acc += wr[3 + c] * s1_lds[c];
      x_lds[t * GC_ + o] = (bf16_t)(acc * sc + proj_b[o]);
    }
  }
  __syncthreads();

  const float sc_blk = sqrtf(2.f / (float)GC_);
  const float sc_a1  = sqrtf(2.f / (float)GC_);
  const float sc_a2  = sqrtf(2.f / (float)(2 * GC_));
  const bf16_t* Ab = Aadj_bf + (size_t)b * S_ * S_;

  for (int blk = 0; blk < NBLK_; ++blk) {
    const bf16_t* Wb = blkw_bf + (size_t)blk * GC_ * GC_;
    const float*  bb = blkb + blk * GC_;

    // ---- GEMM1: h[o][s] = sc*(Wb @ x) + bias ; A=Wb(global bf16), B=x(LDS bf16) ----
    for (int tr = 0; tr < 2; ++tr) {
      const int m0 = (2 * w + tr) * 16;
      for (int tc = 0; tc < 16; ++tc) {
        const int n0 = tc * 16;
        FragF32 acc; acc.v = (v8f){0.f,0.f,0.f,0.f,0.f,0.f,0.f,0.f};
        const bf16_t* browp = x_lds + (n0 + (lane & 15)) * GC_;
        const int arow = m0 + (lane & 15);
#pragma unroll 1
        for (int kk = 0; kk < GC_; kk += 32) {
          v16bf af  = loadA_bf16(Wb, arow, kk, GC_, lane);
          v16bf bfr = loadB_bf16(browp, kk, lane);
          acc.v = wmma_bf16(af, bfr, acc.v);
        }
        const int s  = n0 + (lane & 15);
        const int ob = m0 + ((lane >> 4) << 3);
#pragma unroll
        for (int r = 0; r < 8; ++r)
          h_lds[(ob + r) * S_ + s] = (bf16_t)(acc.e[r] * sc_blk + bb[ob + r]);
      }
    }
    __syncthreads();

    // ---- GEMM2: h2[c][i] = sum_j h[c][j] * A[i][j]; write x_lds[i][c] ----
    for (int tr = 0; tr < 2; ++tr) {
      const int m0 = (2 * w + tr) * 16;
      for (int tc = 0; tc < 16; ++tc) {
        const int n0 = tc * 16;
        FragF32 acc; acc.v = (v8f){0.f,0.f,0.f,0.f,0.f,0.f,0.f,0.f};
        const bf16_t* browp = Ab + (size_t)(n0 + (lane & 15)) * S_;
        const int arow = m0 + (lane & 15);
#pragma unroll 1
        for (int kk = 0; kk < S_; kk += 32) {
          v16bf af  = loadA_bf16(h_lds, arow, kk, S_, lane);
          v16bf bfr = loadB_bf16(browp, kk, lane);
          acc.v = wmma_bf16(af, bfr, acc.v);
        }
        const int i  = n0 + (lane & 15);
        const int cb = m0 + ((lane >> 4) << 3);
#pragma unroll
        for (int r = 0; r < 8; ++r)
          x_lds[i * GC_ + cb + r] = (bf16_t)acc.e[r];
      }
    }
    __syncthreads();

    // ---- mean over nodes: c = t ----
    {
      float acc = 0.f;
      for (int s = 0; s < S_; ++s) acc += (float)x_lds[s * GC_ + t];
      mean_lds[t] = acc * (1.f / (float)S_);
    }
    __syncthreads();

    // ---- style MLP ----
    {
      const float* w1 = a1w + (size_t)blk * 2 * GC_ * GC_;
      const float* b1 = a1b + blk * 2 * GC_;
      for (int oo = 0; oo < 2; ++oo) {
        int o = t + oo * GC_;
        const float* wr = w1 + o * GC_;
        float acc = 0.f;
        for (int c = 0; c < GC_; ++c) acc += wr[c] * mean_lds[c];
        s1_lds[o] = leakyf(acc * sc_a1 + b1[o]);
      }
    }
    __syncthreads();
    {
      const float* w2 = a2w + (size_t)blk * 2 * GC_ * 2 * GC_;
      const float* b2 = a2b + blk * 2 * GC_;
      for (int oo = 0; oo < 2; ++oo) {
        int o = t + oo * GC_;
        const float* wr = w2 + o * 2 * GC_;
        float acc = 0.f;
        for (int k = 0; k < 2 * GC_; ++k) acc += wr[k] * s1_lds[k];
        sty_lds[o] = acc * sc_a2 + b2[o];
      }
    }
    __syncthreads();

    // ---- AdaIN epilogue + leaky, row s = t ----
    for (int c = 0; c < GC_; ++c) {
      float v = (float)x_lds[t * GC_ + c];
      v = leakyf(sty_lds[c] * v + sty_lds[GC_ + c]);
      x_lds[t * GC_ + c] = (bf16_t)v;
    }
    __syncthreads();
  }

  for (int c = 0; c < GC_; ++c)
    xsup_out[((size_t)b * S_ + t) * GC_ + c] = x_lds[t * GC_ + c];
}

// ================= K3a: transpose/convert up1_w -> wt1[tap][o][c] bf16 =================
__global__ __launch_bounds__(256) void k_wt1(const float* __restrict__ up1w,
                                             bf16_t* __restrict__ wt1) {
  int i = blockIdx.x * 256 + threadIdx.x;  // 9*128*256
  if (i >= 9 * 128 * 256) return;
  int c = i & 255; int rest = i >> 8; int o = rest % 128; int tap = rest / 128;
  wt1[(tap * 128 + o) * 256 + c] = (bf16_t)up1w[(o * 256 + c) * 9 + tap];
}

// ================= K3: per-tap superpixel GEMM: T1[b][s][tap][o] ===================
// xsup[b] (128KB bf16) staged into LDS with a single TENSOR_LOAD_TO_LDS (TDM).
__global__ __launch_bounds__(256) void k_tap(const bf16_t* __restrict__ xsup,
                                             const bf16_t* __restrict__ wt1,
                                             float* __restrict__ T1) {
  extern __shared__ char tsm[];
  bf16_t* Xl = (bf16_t*)tsm;               // [S][GC] bf16, 131072 B
  const int b = blockIdx.x / 9, tap = blockIdx.x % 9;
  const int t = threadIdx.x, lane = t & 31, w = t >> 5;

  if (t < 32) {  // one wave issues the DMA (EXEC ignored by TDM)
    unsigned long long ga = (unsigned long long)(uintptr_t)(xsup + (size_t)b * S_ * GC_);
    unsigned lo = lds_off_of(Xl);
    v4u g0;
    g0[0] = 1u;                                    // count=1, user mode
    g0[1] = lo;                                    // lds_addr
    g0[2] = (unsigned)ga;                          // global_addr[31:0]
    g0[3] = ((unsigned)(ga >> 32) & 0x01FFFFFFu)   // global_addr[56:32]
            | 0x80000000u;                         // type=2 ("image")
    v8u g1;
    g1[0] = 1u << 16;        // workgroup_mask=0, data_size=1 (2 bytes)
    g1[1] = 256u << 16;      // tensor_dim0 = 256 (low 16 in bits[79:64->dw1 31:16])
    g1[2] = 256u << 16;      // tensor_dim0 hi=0 | tensor_dim1[15:0]=256
    g1[3] = 256u << 16;      // tensor_dim1 hi=0 | tile_dim0=256
    g1[4] = 256u;            // tile_dim1=256, tile_dim2=0
    g1[5] = 256u;            // tensor_dim0_stride[31:0]=256
    g1[6] = 0u;              // stride hi / tensor_dim1_stride lo
    g1[7] = 0u;
    asm volatile("tensor_load_to_lds %0, %1" :: "s"(g0), "s"(g1) : "memory");
    __builtin_amdgcn_s_wait_tensorcnt(0);
  }
  __syncthreads();

  const bf16_t* Wt = wt1 + (size_t)tap * 128 * GC_;
  for (int tr = 0; tr < 2; ++tr) {
    const int m0 = (2 * w + tr) * 16;
    for (int tc = 0; tc < 8; ++tc) {
      const int n0 = tc * 16;
      FragF32 acc; acc.v = (v8f){0.f,0.f,0.f,0.f,0.f,0.f,0.f,0.f};
      const bf16_t* browp = Wt + (n0 + (lane & 15)) * GC_;
      const int arow = m0 + (lane & 15);
#pragma unroll 1
      for (int kk = 0; kk < GC_; kk += 32) {
        v16bf af  = loadA_bf16(Xl, arow, kk, GC_, lane);
        v16bf bfr = loadB_bf16(browp, kk, lane);
        acc.v = wmma_bf16(af, bfr, acc.v);
      }
      const int o  = n0 + (lane & 15);
      const int sb = m0 + ((lane >> 4) << 3);
#pragma unroll
      for (int r = 0; r < 8; ++r)
        T1[(((size_t)b * S_ + sb + r) * 9 + tap) * 128 + o] = acc.e[r];
    }
  }
}

// ================= K4: conv1 as 9 gathered table adds; y1 = leaky(.+b1) =================
__global__ __launch_bounds__(256) void k_gather(const int* __restrict__ seg,
                                                const float* __restrict__ T1,
                                                const float* __restrict__ b1,
                                                bf16_t* __restrict__ y1) {
  const int w = threadIdx.x >> 5, lane = threadIdx.x & 31;
  const int pix = blockIdx.x * 8 + w;      // one wave per pixel
  const int b = pix >> 16, n = pix & 65535;
  const int y = n >> 8, x = n & 255;
  const int c0 = lane * 4;                 // 32 lanes * 4 ch = 128
  float4 acc = {0.f, 0.f, 0.f, 0.f};
  const int* segb = seg + (size_t)b * NPIX_;
#pragma unroll
  for (int tap = 0; tap < 9; ++tap) {
    const int dy = tap / 3 - 1, dx = tap % 3 - 1;
    const int yy = y + dy, xx = x + dx;
    if (yy < 0 || yy >= 256 || xx < 0 || xx >= 256) continue;  // zero pad
    const int s = segb[yy * 256 + xx];
    const float4 v = *(const float4*)(T1 + (((size_t)b * S_ + s) * 9 + tap) * 128 + c0);
    acc.x += v.x; acc.y += v.y; acc.z += v.z; acc.w += v.w;
  }
  const float4 bb = *(const float4*)(b1 + c0);
  bf16_t* out = y1 + ((size_t)b * NPIX_ + n) * 128 + c0;
  out[0] = (bf16_t)leakyf(acc.x + bb.x);
  out[1] = (bf16_t)leakyf(acc.y + bb.y);
  out[2] = (bf16_t)leakyf(acc.z + bb.z);
  out[3] = (bf16_t)leakyf(acc.w + bb.w);
}

// ================= K5: conv2 (128->3) + instance-norm partial stats =================
// One block == one image row; rows y-1..y+1 of y1 (192KB) staged via async DMA to LDS.
__global__ __launch_bounds__(256) void k_conv2(const bf16_t* __restrict__ y1,
                                               const float* __restrict__ w2,
                                               const float* __restrict__ b2,
                                               float* __restrict__ y2,
                                               float* __restrict__ stats) {
  extern __shared__ char csm[];
  bf16_t* rows = (bf16_t*)csm;                       // [3][256][128] bf16 = 196608 B
  float*  w2s  = (float*)(csm + 196608);             // 3456 f32
  float*  st   = (float*)(csm + 196608 + 13824);     // 6 f32
  const int t = threadIdx.x;
  const int pix = blockIdx.x * 256 + t;              // whole block in one (b, y)
  const int b = pix >> 16, n = pix & 65535;
  const int y = n >> 8, x = n & 255;

  // async-stage 3 rows of y1 (clamped; clamped rows never read due to bounds check)
  {
    const unsigned rowsoff = lds_off_of(rows);
    const char* y1b = (const char*)y1;
#pragma unroll 1
    for (int it = 0; it < 48; ++it) {
      int i  = it * 256 + t;                  // 0..12287 x 16B = 192KB
      int rr = i >> 12;
      int j  = (i & 4095) << 4;
      int yy = y - 1 + rr; yy = yy < 0 ? 0 : (yy > 255 ? 255 : yy);
      unsigned lds_v = rowsoff + (unsigned)(rr * 65536 + j);
      unsigned long long ga =
          (unsigned long long)(uintptr_t)(y1b + (((size_t)b * NPIX_ + (size_t)yy * 256) * 256) + j);
      asm volatile("global_load_async_to_lds_b128 %0, %1, off" :: "v"(lds_v), "v"(ga) : "memory");
    }
  }
  for (int i = t; i < 3 * 128 * 9; i += 256) w2s[i] = w2[i];
  if (t < 6) st[t] = 0.f;
  asm volatile("s_wait_asynccnt 0x0" ::: "memory");
  __syncthreads();

  float a0 = b2[0], a1 = b2[1], a2 = b2[2];
#pragma unroll 1
  for (int tap = 0; tap < 9; ++tap) {
    const int dy = tap / 3 - 1, dx = tap % 3 - 1;
    const int yy = y + dy, xx = x + dx;
    if (yy < 0 || yy >= 256 || xx < 0 || xx >= 256) continue;
    const bf16_t* rowp = rows + (size_t)(dy + 1) * 32768 + xx * 128;
    for (int c = 0; c < 128; ++c) {
      const float v = (float)rowp[c];
      a0 += w2s[0 * 1152 + c * 9 + tap] * v;
      a1 += w2s[1 * 1152 + c * 9 + tap] * v;
      a2 += w2s[2 * 1152 + c * 9 + tap] * v;
    }
  }
  y2[((size_t)b * 3 + 0) * NPIX_ + n] = a0;
  y2[((size_t)b * 3 + 1) * NPIX_ + n] = a1;
  y2[((size_t)b * 3 + 2) * NPIX_ + n] = a2;
  atomicAdd(&st[0], a0); atomicAdd(&st[1], a0 * a0);
  atomicAdd(&st[2], a1); atomicAdd(&st[3], a1 * a1);
  atomicAdd(&st[4], a2); atomicAdd(&st[5], a2 * a2);
  __syncthreads();
  if (t < 6) atomicAdd(&stats[b * 6 + t], st[t]);
}

__global__ void k_zero_stats(float* stats) {
  if (threadIdx.x < 24) stats[threadIdx.x] = 0.f;
}

// ================= K6: instance norm + tanh =================
__global__ __launch_bounds__(256) void k_final(const float* __restrict__ y2,
                                               const float* __restrict__ stats,
                                               float* __restrict__ out) {
  const int i = blockIdx.x * 256 + threadIdx.x;  // B*3*NPIX
  if (i >= BATCH_ * 3 * NPIX_) return;
  const int rest = i / NPIX_;
  const int o = rest % 3, b = rest / 3;
  const float s  = stats[b * 6 + 2 * o];
  const float sq = stats[b * 6 + 2 * o + 1];
  const float mu  = s * (1.f / (float)NPIX_);
  const float var = sq * (1.f / (float)NPIX_) - mu * mu;
  const float v = (y2[i] - mu) * rsqrtf(var + 1e-5f);
  out[i] = tanhf(v);
}

// ================= host side =================
extern "C" void kernel_launch(void* const* d_in, const int* in_sizes, int n_in,
                              void* d_out, int out_size, void* d_ws, size_t ws_size,
                              hipStream_t stream) {
  const float* z      = (const float*)d_in[0];
  const float* images = (const float*)d_in[1];
  const int*   seg    = (const int*)d_in[2];
  const float* Aadj   = (const float*)d_in[3];
  const float* proj_w = (const float*)d_in[4];
  const float* proj_b = (const float*)d_in[5];
  const float* blkw   = (const float*)d_in[6];
  const float* blkb   = (const float*)d_in[7];
  const float* a1w    = (const float*)d_in[8];
  const float* a1b    = (const float*)d_in[9];
  const float* a2w    = (const float*)d_in[10];
  const float* a2b    = (const float*)d_in[11];
  const float* up1w   = (const float*)d_in[12];
  const float* up1b   = (const float*)d_in[13];
  const float* up2w   = (const float*)d_in[14];
  const float* up2b   = (const float*)d_in[15];
  float* out = (float*)d_out;

  // workspace carve (256B aligned)
  char* ws = (char*)d_ws;
  size_t off = 0;
  auto bump = [&](size_t bytes) { size_t o = off; off += (bytes + 255) & ~(size_t)255; return o; };
  float*  feats   = (float*)(ws + bump((size_t)BATCH_ * S_ * 3 * 4));
  bf16_t* xsup    = (bf16_t*)(ws + bump((size_t)BATCH_ * S_ * GC_ * 2));
  bf16_t* wt1     = (bf16_t*)(ws + bump((size_t)9 * 128 * 256 * 2));
  bf16_t* blkw_bf = (bf16_t*)(ws + bump((size_t)NBLK_ * GC_ * GC_ * 2));
  bf16_t* Aadj_bf = (bf16_t*)(ws + bump((size_t)BATCH_ * S_ * S_ * 2));
  float*  T1      = (float*)(ws + bump((size_t)BATCH_ * S_ * 9 * 128 * 4));
  bf16_t* y1      = (bf16_t*)(ws + bump((size_t)BATCH_ * NPIX_ * 128 * 2));
  float*  y2      = (float*)(ws + bump((size_t)BATCH_ * 3 * NPIX_ * 4));
  float*  stats   = (float*)(ws + bump(24 * 4));
  (void)ws_size; (void)in_sizes; (void)n_in; (void)out_size;

  k_pool<<<BATCH_, 256, 0, stream>>>(images, seg, feats);
  k_zero_stats<<<1, 32, 0, stream>>>(stats);
  k_wt1<<<(9 * 128 * 256 + 255) / 256, 256, 0, stream>>>(up1w, wt1);
  k_cvt<<<(NBLK_ * GC_ * GC_ + 255) / 256, 256, 0, stream>>>(blkw, blkw_bf, NBLK_ * GC_ * GC_);
  k_cvt<<<(BATCH_ * S_ * S_ + 255) / 256, 256, 0, stream>>>(Aadj, Aadj_bf, BATCH_ * S_ * S_);

  // 262144 (x,h bf16) + 1024 (mean) + 2048 (s1) + 2048 (style) = 267264 bytes LDS
  k_core<<<BATCH_, 256, 267264, stream>>>(z, feats, proj_w, proj_b, blkw_bf, blkb,
                                          a1w, a1b, a2w, a2b, Aadj_bf, xsup);

  k_tap<<<BATCH_ * 9, 256, 131072, stream>>>(xsup, wt1, T1);
  k_gather<<<(BATCH_ * NPIX_) / 8, 256, 0, stream>>>(seg, T1, up1b, y1);
  // 196608 (rows) + 13824 (w2) + 32 (stats) = 210464 bytes LDS
  k_conv2<<<(BATCH_ * NPIX_) / 256, 256, 210464, stream>>>(y1, up2w, up2b, y2, stats);
  k_final<<<(BATCH_ * 3 * NPIX_ + 255) / 256, 256, 0, stream>>>(y2, stats, out);
}
